// MultiHeadAttention_Homo_68676527063541
// MI455X (gfx1250) — matrix operand order repaired and checked by reading, not compiled, not run on hardware
//
#include <hip/hip_runtime.h>
#include <hip/hip_bf16.h>
#include <math.h>

// ---------------------------------------------------------------------------
// Graph-transformer layer for MI455X (gfx1250, wave32, WMMA).
//   prep:     convert 5 weight mats f32[K][N] -> f16 transposed [N][K]
//   proj:     q|k|v|ffn = feats @ [Wq|Wk|Wv|Wffn] + bias   (WMMA f16->f32)
//   edge:     per-edge per-head softmax-numerator scatter   (f32 atomics, L2)
//   finalize: o = (wv/z) @ Wo + bo ; h = ffn + o ; out = h + LN(h)
// ---------------------------------------------------------------------------

typedef __attribute__((ext_vector_type(16))) _Float16 v16h;
typedef __attribute__((ext_vector_type(8)))  _Float16 v8h;
typedef __attribute__((ext_vector_type(8)))  float    v8f;

#define IN_FEATS 128
#define O_HEAD   128
#define NUM_HEAD 8
#define LN_EPS   1e-5f
#define INV_SQRT_IN 0.08838834764831845f   /* 1/sqrt(128) */

// ---- WMMA fragment loaders (CDNA5 ISA 7.12.2 layouts) ----------------------

// A: 16x32 f16 from row-major f32 source (converted in-flight).
// Lane L (L<16): M=L, halves = K{0..7,16..23}; lane>=16: M=L-16,
// halves = K{8..15,24..31}.
__device__ __forceinline__ v16h load_A_frag(const float* __restrict__ A,
                                            int lda, int k0) {
  const int lane = threadIdx.x & 31;
  const int m    = lane & 15;
  const int kb   = (lane >> 4) * 8;
  const float* p = A + (size_t)m * lda + k0;
  v16h a;
#pragma unroll
  for (int j = 0; j < 8; ++j) a[j] = (_Float16)p[kb + j];
#pragma unroll
  for (int j = 0; j < 8; ++j) a[8 + j] = (_Float16)p[16 + kb + j];
  return a;
}

// B: 32x16 f16 from TRANSPOSED f16 weights Wt[N][K] (K=128 contiguous).
// Lane L holds column N=L&15; lanes 0-15 K={0..7,16..23}, lanes 16-31 K={8..15,24..31}.
// Two runs of 8 halves -> two global_load_b128.
__device__ __forceinline__ v16h load_B_T(const _Float16* __restrict__ Wt,
                                         int col0, int k0) {
  const int lane = threadIdx.x & 31;
  const int n    = lane & 15;
  const int kb   = (lane >> 4) * 8;
  const _Float16* p = Wt + (size_t)(col0 + n) * IN_FEATS + k0 + kb;
  v8h lo = *(const v8h*)p;
  v8h hi = *(const v8h*)(p + 16);
  return __builtin_shufflevector(lo, hi, 0, 1, 2, 3, 4, 5, 6, 7,
                                 8, 9, 10, 11, 12, 13, 14, 15);
}

// ---- Kernel 0: weight prep (f32 row-major -> f16 transposed) ---------------
// Wt layout: mat-major, Wt[mat][n][k] = W_mat[k][n].  mats: q,k,v,ffn,o
__global__ __launch_bounds__(256)
void prep_weights(const float* __restrict__ Wq, const float* __restrict__ Wk,
                  const float* __restrict__ Wv, const float* __restrict__ Wf,
                  const float* __restrict__ Wo, _Float16* __restrict__ Wt) {
  const int i = blockIdx.x * 256 + threadIdx.x;     // 5 * 128 * 128 elements
  if (i >= 5 * IN_FEATS * O_HEAD) return;
  const int mat = i >> 14;
  const int n   = (i >> 7) & 127;
  const int k   = i & 127;
  const float* W = (mat == 0) ? Wq : (mat == 1) ? Wk : (mat == 2) ? Wv
                 : (mat == 3) ? Wf : Wo;
  Wt[i] = (_Float16)W[(size_t)k * O_HEAD + n];
}

// ---- Kernel 1: fused projections  [N,128] x [128, 4*128] -------------------
// One wave per (16-row tile, matrix). A-fragments for all 4 K-steps are kept
// in registers and reused across the 8 column tiles of that matrix.
__global__ __launch_bounds__(256)
void proj_kernel(const float* __restrict__ feats,
                 const _Float16* __restrict__ Wt,
                 const float* __restrict__ bq, const float* __restrict__ bk,
                 const float* __restrict__ bv, const float* __restrict__ bf,
                 float* __restrict__ q, float* __restrict__ k,
                 float* __restrict__ v, float* __restrict__ ffn,
                 int n) {
  const int wid     = blockIdx.x * 8 + (threadIdx.x >> 5);
  const int nRow    = (n + 15) >> 4;
  const int rowTile = wid >> 2;          // 4 matrices per row tile
  const int mat     = wid & 3;           // 0:q 1:k 2:v 3:ffn
  if (rowTile >= nRow) return;
  const int row0 = rowTile * 16;
  const bool fullTile = (row0 + 16 <= n);

  const _Float16* W = Wt + (size_t)mat * IN_FEATS * O_HEAD;
  const float* bias; float* out;
  switch (mat) {
    case 0:  bias = bq; out = q;   break;
    case 1:  bias = bk; out = k;   break;
    case 2:  bias = bv; out = v;   break;
    default: bias = bf; out = ffn; break;
  }

  // load + convert the 16x128 A panel once (4 K-step fragments)
  const float* Abase = feats + (size_t)row0 * IN_FEATS;
  v16h a[4];
#pragma unroll
  for (int ks = 0; ks < 4; ++ks) a[ks] = load_A_frag(Abase, IN_FEATS, ks * 32);

  const int lane  = threadIdx.x & 31;
  const int nIdx  = lane & 15;
  const int rbase = (lane >> 4) * 8;

#pragma unroll
  for (int ct = 0; ct < 8; ++ct) {
    const int col0 = ct * 16;
    v8f c = {};
#pragma unroll
    for (int ks = 0; ks < 4; ++ks) {
      v16h b = load_B_T(W, col0, ks * 32);
      c = __builtin_amdgcn_wmma_f32_16x16x32_f16(false, a[ks], false, b,
                                                 (short)0, c, false, false);
    }
    // D layout: lanes 0-15 -> M=r, N=lane; lanes 16-31 -> M=r+8, N=lane-16.
    const float bb = bias[col0 + nIdx];
    float* op = out + (size_t)(row0 + rbase) * O_HEAD + col0 + nIdx;
    if (fullTile) {       // straight-line stores, immediate offsets
#pragma unroll
      for (int r = 0; r < 8; ++r) op[(size_t)r * O_HEAD] = c[r] + bb;
    } else {
#pragma unroll
      for (int r = 0; r < 8; ++r)
        if (row0 + rbase + r < n) op[(size_t)r * O_HEAD] = c[r] + bb;
    }
  }
}

// ---- Kernel 2: edge scatter (one wave per edge) ----------------------------
__global__ __launch_bounds__(256)
void edge_kernel(const int* __restrict__ esrc, const int* __restrict__ edst,
                 const float* __restrict__ q, const float* __restrict__ k,
                 const float* __restrict__ v,
                 float* __restrict__ wv, float* __restrict__ z, int nE) {
  const int e = (int)((blockIdx.x * (size_t)blockDim.x + threadIdx.x) >> 5);
  if (e >= nE) return;
  const int lane = threadIdx.x & 31;          // lane covers feats 4l..4l+3
  const int src  = esrc[e];
  const int dst  = edst[e];

  const float4 k4 = *(const float4*)(k + (size_t)src * O_HEAD + lane * 4);
  const float4 q4 = *(const float4*)(q + (size_t)dst * O_HEAD + lane * 4);
  float p = k4.x * q4.x + k4.y * q4.y + k4.z * q4.z + k4.w * q4.w;
  // reduce across the 4 lanes of this head (16 feats / 4 per lane)
  p += __shfl_xor(p, 1, 32);
  p += __shfl_xor(p, 2, 32);
  float s = fminf(5.0f, fmaxf(-5.0f, p * INV_SQRT_IN));
  s = expf(s);

  const float4 v4 = *(const float4*)(v + (size_t)src * O_HEAD + lane * 4);
  float* wd = wv + (size_t)dst * O_HEAD + lane * 4;
  unsafeAtomicAdd(wd + 0, v4.x * s);
  unsafeAtomicAdd(wd + 1, v4.y * s);
  unsafeAtomicAdd(wd + 2, v4.z * s);
  unsafeAtomicAdd(wd + 3, v4.w * s);
  if ((lane & 3) == 0)
    unsafeAtomicAdd(z + (size_t)dst * NUM_HEAD + (lane >> 2), s);
}

// ---- Kernel 3: o = (wv/z) @ Wo + bo ; h = ffn + o ; out = h + LN(h) --------
// Block = 16 nodes, 256 threads = 8 waves; wave w computes cols [16w,16w+16).
__global__ __launch_bounds__(256)
void finalize_kernel(const float* __restrict__ wv, const float* __restrict__ z,
                     const float* __restrict__ ffn,
                     const _Float16* __restrict__ WoT,
                     const float* __restrict__ bo,
                     const float* __restrict__ ln_g, const float* __restrict__ ln_b,
                     float* __restrict__ out, int n) {
  __shared__ float hA[16 * O_HEAD];   // normalized attention rows
  __shared__ float hH[16 * O_HEAD];   // h = ffn + o
  const int row0 = blockIdx.x * 16;
  const int t    = threadIdx.x;
  const bool fullTile = (row0 + 16 <= n);

  // normalize wv by z into LDS (each thread: one row, 8 feats)
  {
    const int r = t >> 4, f0 = (t & 15) * 8;
    const int row = fullTile ? (row0 + r) : min(row0 + r, n - 1);
    const float* wr = wv + (size_t)row * O_HEAD;
    const float* zr = z + (size_t)row * NUM_HEAD;
#pragma unroll
    for (int j = 0; j < 8; ++j) {
      const int f = f0 + j;
      hA[r * O_HEAD + f] = wr[f] / zr[f >> 4];
    }
  }
  __syncthreads();

  const int wave = t >> 5, lane = t & 31;
  const int col0 = wave * 16;
  v8f c = {};
#pragma unroll
  for (int ks = 0; ks < 4; ++ks) {
    // A fragment straight out of LDS (converted to f16)
    const int m  = lane & 15;
    const int kb = (lane >> 4) * 8;
    const float* p = hA + m * O_HEAD + ks * 32;
    v16h a;
#pragma unroll
    for (int j = 0; j < 8; ++j) a[j] = (_Float16)p[kb + j];
#pragma unroll
    for (int j = 0; j < 8; ++j) a[8 + j] = (_Float16)p[16 + kb + j];
    v16h b = load_B_T(WoT, col0, ks * 32);
    c = __builtin_amdgcn_wmma_f32_16x16x32_f16(false, a, false, b,
                                               (short)0, c, false, false);
  }

  const int nIdx  = lane & 15;
  const int rbase = (lane >> 4) * 8;
  const float bb  = bo[col0 + nIdx];
  const float* fp = ffn + (size_t)(row0 + rbase) * O_HEAD + col0 + nIdx;
  if (fullTile) {
#pragma unroll
    for (int r = 0; r < 8; ++r)
      hH[(rbase + r) * O_HEAD + col0 + nIdx] = c[r] + bb + fp[(size_t)r * O_HEAD];
  } else {
#pragma unroll
    for (int r = 0; r < 8; ++r) {
      const int row = min(row0 + rbase + r, n - 1);
      hH[(rbase + r) * O_HEAD + col0 + nIdx] =
          c[r] + bb + ffn[(size_t)row * O_HEAD + col0 + nIdx];
    }
  }
  __syncthreads();

  // LayerNorm over 128 feats: 16 threads per row, 8 feats per thread.
  const int row = t >> 4, l16 = t & 15;
  const float* hr = hH + row * O_HEAD;
  float s = 0.f;
#pragma unroll
  for (int j = 0; j < 8; ++j) s += hr[l16 * 8 + j];
  s += __shfl_xor(s, 1, 32); s += __shfl_xor(s, 2, 32);
  s += __shfl_xor(s, 4, 32); s += __shfl_xor(s, 8, 32);
  const float mu = s * (1.0f / O_HEAD);
  float vs = 0.f;
#pragma unroll
  for (int j = 0; j < 8; ++j) {
    const float d = hr[l16 * 8 + j] - mu;
    vs += d * d;
  }
  vs += __shfl_xor(vs, 1, 32); vs += __shfl_xor(vs, 2, 32);
  vs += __shfl_xor(vs, 4, 32); vs += __shfl_xor(vs, 8, 32);
  const float rstd = rsqrtf(vs * (1.0f / O_HEAD) + LN_EPS);

  const int orow = row0 + row;
  if (orow < n) {
    float* op = out + (size_t)orow * O_HEAD;
#pragma unroll
    for (int j = 0; j < 8; ++j) {
      const int f = l16 * 8 + j;
      const float h = hr[f];
      op[f] = h + (h - mu) * rstd * ln_g[f] + ln_b[f];
    }
  }
}

// ---------------------------------------------------------------------------
extern "C" void kernel_launch(void* const* d_in, const int* in_sizes, int n_in,
                              void* d_out, int out_size, void* d_ws, size_t ws_size,
                              hipStream_t stream) {
  const float* feats = (const float*)d_in[0];
  const int*   esrc  = (const int*)d_in[1];
  const int*   edst  = (const int*)d_in[2];
  const float* Wq    = (const float*)d_in[3];
  const float* bq    = (const float*)d_in[4];
  const float* Wk    = (const float*)d_in[5];
  const float* bk    = (const float*)d_in[6];
  const float* Wv    = (const float*)d_in[7];
  const float* bv    = (const float*)d_in[8];
  const float* Wo    = (const float*)d_in[9];
  const float* bo    = (const float*)d_in[10];
  const float* Wf    = (const float*)d_in[11];
  const float* bf    = (const float*)d_in[12];
  const float* ln_g  = (const float*)d_in[13];
  const float* ln_b  = (const float*)d_in[14];
  float* out = (float*)d_out;

  const int N = in_sizes[0] / IN_FEATS;
  const int E = in_sizes[1];

  float* ws  = (float*)d_ws;
  float* q   = ws;
  float* k   = q   + (size_t)N * O_HEAD;
  float* v   = k   + (size_t)N * O_HEAD;
  float* ffn = v   + (size_t)N * O_HEAD;
  float* wv  = ffn + (size_t)N * O_HEAD;
  float* z   = wv  + (size_t)N * O_HEAD;
  _Float16* Wt = (_Float16*)(z + (size_t)N * NUM_HEAD);  // 5 * 128*128 f16

  // zero accumulators (graph-capture safe)
  hipMemsetAsync(wv, 0, (size_t)N * (O_HEAD + NUM_HEAD) * sizeof(float), stream);

  // weight prep: 5 * 16384 elements
  prep_weights<<<(5 * IN_FEATS * O_HEAD + 255) / 256, 256, 0, stream>>>(
      Wq, Wk, Wv, Wf, Wo, Wt);

  // projections: one wave per (row tile, matrix); 8 waves per block
  const int nRowTiles = (N + 15) / 16;
  const int nWaves    = nRowTiles * 4;
  proj_kernel<<<(nWaves + 7) / 8, 256, 0, stream>>>(
      feats, Wt, bq, bk, bv, bf, q, k, v, ffn, N);

  // edges: one wave per edge, 8 waves per block
  edge_kernel<<<(E + 7) / 8, 256, 0, stream>>>(esrc, edst, q, k, v, wv, z, E);

  // finalize: one block per 16 nodes
  finalize_kernel<<<nRowTiles, 256, 0, stream>>>(
      wv, z, ffn, Wt + (size_t)4 * IN_FEATS * O_HEAD, bo, ln_g, ln_b, out, N);
}
